// GCNNet_55920474193965
// MI455X (gfx1250) — matrix operand order repaired and checked by reading, not compiled, not run on hardware
//
#include <hip/hip_runtime.h>

typedef __attribute__((ext_vector_type(16))) __bf16 v16bf;
typedef __attribute__((ext_vector_type(8)))  float  v8f;

// round-to-nearest-even fp32 -> bf16
__device__ __forceinline__ unsigned short bf16_rn(float f) {
    unsigned int u = __float_as_uint(f);
    u += 0x7FFFu + ((u >> 16) & 1u);
    return (unsigned short)(u >> 16);
}

union V16 { v16bf v; unsigned int u[8]; };

// ---------------------------------------------------------------------------
// Split-bf16 GEMM: C[M x NC] = A[M x K] * B[K x NC], fp32 in/out,
// 3x v_wmma_f32_16x16x32_bf16 per 16x16x32 step (hi*hi + hi*lo + lo*hi).
// Block: 256 threads = 8 waves; 128-row M tile; full NC kept in accumulators.
// ---------------------------------------------------------------------------
template <int K, int NC>
__global__ __launch_bounds__(256) void gemm_bf16x3(
    const float* __restrict__ A, const float* __restrict__ B,
    float* __restrict__ C, int M)
{
    constexpr int BM   = 128;
    constexpr int KT   = 32;
    constexpr int ASTR = KT + 2;          // 34 ushorts -> 17 dwords: bank-conflict-free
    constexpr int NSUB = NC / 16;

    __shared__ unsigned short sAhi[BM * ASTR];
    __shared__ unsigned short sAlo[BM * ASTR];
    __shared__ unsigned short sBhi[NC * ASTR];   // stored transposed: [col][k]
    __shared__ unsigned short sBlo[NC * ASTR];

    const int tid   = threadIdx.x;
    const int lane  = tid & 31;
    const int wave  = tid >> 5;
    const int half  = lane >> 4;
    const int rrow  = lane & 15;
    const int mbase = blockIdx.x * BM;

    v8f acc[NSUB];
#pragma unroll
    for (int n = 0; n < NSUB; ++n)
        acc[n] = (v8f){0.f, 0.f, 0.f, 0.f, 0.f, 0.f, 0.f, 0.f};

    for (int k0 = 0; k0 < K; k0 += KT) {
        // ---- stage A tile (BM x KT fp32) into LDS as bf16 hi/lo ----
#pragma unroll
        for (int p = 0; p < (BM * KT) / (256 * 4); ++p) {
            int idx  = p * 256 + tid;          // float4 index
            int r    = idx >> 3;               // 8 float4 per row of 32
            int c4   = idx & 7;
            int grow = mbase + r;
            float4 v = make_float4(0.f, 0.f, 0.f, 0.f);
            if (grow < M)
                v = *(const float4*)(A + (size_t)grow * K + k0 + c4 * 4);
            float vv[4] = {v.x, v.y, v.z, v.w};
            int base = r * ASTR + c4 * 4;
#pragma unroll
            for (int j = 0; j < 4; ++j) {
                unsigned short h = bf16_rn(vv[j]);
                float hf = __uint_as_float((unsigned int)h << 16);
                sAhi[base + j] = h;
                sAlo[base + j] = bf16_rn(vv[j] - hf);
            }
        }
        // ---- stage B tile (KT x NC fp32) transposed into LDS ----
#pragma unroll
        for (int p = 0; p < (KT * NC) / (256 * 4); ++p) {
            int idx = p * 256 + tid;
            int kk  = idx / (NC / 4);
            int c4  = idx % (NC / 4);
            float4 v = *(const float4*)(B + (size_t)(k0 + kk) * NC + c4 * 4);
            float vv[4] = {v.x, v.y, v.z, v.w};
#pragma unroll
            for (int j = 0; j < 4; ++j) {
                int col = c4 * 4 + j;
                unsigned short h = bf16_rn(vv[j]);
                float hf = __uint_as_float((unsigned int)h << 16);
                sBhi[col * ASTR + kk] = h;
                sBlo[col * ASTR + kk] = bf16_rn(vv[j] - hf);
            }
        }
        __syncthreads();

        // ---- A operand per ISA 16-bit 16x32 layout ----
        V16 ahi, alo;
        {
            const unsigned short* pAh = &sAhi[(wave * 16 + rrow) * ASTR];
            const unsigned short* pAl = &sAlo[(wave * 16 + rrow) * ASTR];
            const int ko = half * 8;
#pragma unroll
            for (int v2 = 0; v2 < 8; ++v2) {
                int kk = ((v2 < 4) ? 0 : 16) + ko + 2 * (v2 & 3);
                ahi.u[v2] = *(const unsigned int*)(pAh + kk);
                alo.u[v2] = *(const unsigned int*)(pAl + kk);
            }
        }
#pragma unroll
        for (int n = 0; n < NSUB; ++n) {
            // ---- B operand per ISA 32x16 layout (K-pairs per dword) ----
            V16 bhi, blo;
            const unsigned short* pBh = &sBhi[(n * 16 + rrow) * ASTR];
            const unsigned short* pBl = &sBlo[(n * 16 + rrow) * ASTR];
#pragma unroll
            for (int v2 = 0; v2 < 8; ++v2) {
                int kk = ((v2 < 4) ? 0 : 16) + half * 8 + 2 * (v2 & 3);
                bhi.u[v2] = *(const unsigned int*)(pBh + kk);
                blo.u[v2] = *(const unsigned int*)(pBl + kk);
            }
            acc[n] = __builtin_amdgcn_wmma_f32_16x16x32_bf16(
                false, ahi.v, false, bhi.v, (short)0, acc[n], false, false);
            acc[n] = __builtin_amdgcn_wmma_f32_16x16x32_bf16(
                false, ahi.v, false, blo.v, (short)0, acc[n], false, false);
            acc[n] = __builtin_amdgcn_wmma_f32_16x16x32_bf16(
                false, alo.v, false, bhi.v, (short)0, acc[n], false, false);
        }
        __syncthreads();
    }

    // ---- store per ISA 32-bit 16x16 C/D layout ----
#pragma unroll
    for (int n = 0; n < NSUB; ++n) {
#pragma unroll
        for (int r = 0; r < 8; ++r) {
            int grow = mbase + wave * 16 + r + 8 * half;
            if (grow < M)
                C[(size_t)grow * NC + n * 16 + rrow] = acc[n][r];
        }
    }
}

// ---------------------------------------------------------------------------
// Helpers: zero / degree / rsqrt
// ---------------------------------------------------------------------------
__global__ void zero_f4(float4* __restrict__ p, int n4) {
    int i = blockIdx.x * blockDim.x + threadIdx.x;
    if (i < n4) p[i] = make_float4(0.f, 0.f, 0.f, 0.f);
}

__global__ void init_one(float* __restrict__ p, int n) {
    int i = blockIdx.x * blockDim.x + threadIdx.x;
    if (i < n) p[i] = 1.0f;   // self-loop contribution to in-degree
}

__global__ void deg_accum(const int* __restrict__ dst, float* __restrict__ deg, int E) {
    int e = blockIdx.x * blockDim.x + threadIdx.x;
    if (e < E) atomicAdd(&deg[dst[e]], 1.0f);
}

__global__ void deg_rsqrt(float* __restrict__ d, int n) {
    int i = blockIdx.x * blockDim.x + threadIdx.x;
    if (i < n) d[i] = __frsqrt_rn(d[i]);
}

// ---------------------------------------------------------------------------
// Edge aggregation: one wave per edge; lane covers F/32 contiguous floats.
// out[dst] += H[src] * dinv[src]*dinv[dst]   (non-returning f32 atomics)
// ---------------------------------------------------------------------------
template <int F>
__global__ __launch_bounds__(256) void agg_edges(
    const float* __restrict__ H, const int* __restrict__ src,
    const int* __restrict__ dst, const float* __restrict__ dinv,
    float* __restrict__ out, int E)
{
    int wv = blockIdx.x * (blockDim.x >> 5) + (threadIdx.x >> 5);
    if (wv >= E) return;
    int lane = threadIdx.x & 31;
    int s = src[wv], d = dst[wv];
    float w = dinv[s] * dinv[d];
    constexpr int PER = F / 32;
    const float* hrow = H + (size_t)s * F;
    float* orow = out + (size_t)d * F;
    if constexpr (PER == 4) {
        float4 hv = ((const float4*)hrow)[lane];
        float* o = orow + lane * 4;
        atomicAdd(o + 0, hv.x * w);
        atomicAdd(o + 1, hv.y * w);
        atomicAdd(o + 2, hv.z * w);
        atomicAdd(o + 3, hv.w * w);
    } else {
        float2 hv = ((const float2*)hrow)[lane];
        float* o = orow + lane * 2;
        atomicAdd(o + 0, hv.x * w);
        atomicAdd(o + 1, hv.y * w);
    }
}

// ---------------------------------------------------------------------------
// Epilogue 1: h = relu(agg + H1 * dinv^2 + b1)   (F = 128, in place over agg)
// ---------------------------------------------------------------------------
__global__ void post1(float* __restrict__ G, const float* __restrict__ H1,
                      const float* __restrict__ dinv, const float* __restrict__ b,
                      int M)
{
    int idx = blockIdx.x * blockDim.x + threadIdx.x;   // float4 index
    if (idx >= M * 32) return;
    int n  = idx >> 5;
    int f4 = idx & 31;
    float di = dinv[n];
    float w  = di * di;
    float4 g  = ((const float4*)G)[idx];
    float4 h  = ((const float4*)H1)[idx];
    float4 bb = ((const float4*)b)[f4];
    float4 r;
    r.x = fmaxf(g.x + h.x * w + bb.x, 0.f);
    r.y = fmaxf(g.y + h.y * w + bb.y, 0.f);
    r.z = fmaxf(g.z + h.z * w + bb.z, 0.f);
    r.w = fmaxf(g.w + h.w * w + bb.w, 0.f);
    ((float4*)G)[idx] = r;
}

// ---------------------------------------------------------------------------
// Epilogue 2: out = agg(out) + H2 * dinv^2 + b2   (F = 64, no relu)
// ---------------------------------------------------------------------------
__global__ void post2(float* __restrict__ O, const float* __restrict__ H2,
                      const float* __restrict__ dinv, const float* __restrict__ b,
                      int M)
{
    int idx = blockIdx.x * blockDim.x + threadIdx.x;   // float4 index
    if (idx >= M * 16) return;
    int n  = idx >> 4;
    int f4 = idx & 15;
    float di = dinv[n];
    float w  = di * di;
    float4 o  = ((const float4*)O)[idx];
    float4 h  = ((const float4*)H2)[idx];
    float4 bb = ((const float4*)b)[f4];
    o.x += h.x * w + bb.x;
    o.y += h.y * w + bb.y;
    o.z += h.z * w + bb.z;
    o.w += h.w * w + bb.w;
    ((float4*)O)[idx] = o;
}

// ---------------------------------------------------------------------------
extern "C" void kernel_launch(void* const* d_in, const int* in_sizes, int n_in,
                              void* d_out, int out_size, void* d_ws, size_t ws_size,
                              hipStream_t stream)
{
    const float* x  = (const float*)d_in[0];
    const int*   ei = (const int*)d_in[1];   // int32 (JAX x64 disabled)
    const float* W1 = (const float*)d_in[2];
    const float* b1 = (const float*)d_in[3];
    const float* W2 = (const float*)d_in[4];
    const float* b2 = (const float*)d_in[5];
    float* out = (float*)d_out;

    constexpr int IN = 512, HID = 128, NCLS = 64;
    const int M = in_sizes[0] / IN;     // 100000 nodes
    const int E = in_sizes[1] / 2;      // 1.6M edges (src then dst)
    const int* src = ei;
    const int* dst = ei + E;

    // workspace layout (floats): dinv | H1 (aliased by H2) | G1 (agg1 -> h)
    float* ws   = (float*)d_ws;
    float* dinv = ws;
    size_t off  = ((size_t)M + 255) & ~(size_t)255;
    float* H1   = ws + off;
    float* G1   = H1 + (size_t)M * HID;
    float* H2   = H1;                   // safe: H1 dead after post1

    const int T = 256;
    auto cdiv = [](long long a, long long b) { return (int)((a + b - 1) / b); };

    // degrees (init 1.0 for self loop) -> dinv = rsqrt(deg)
    init_one<<<cdiv(M, T), T, 0, stream>>>(dinv, M);
    // zero aggregation targets (d_out is poisoned; atomics need zeros)
    zero_f4<<<cdiv((size_t)M * HID / 4, T), T, 0, stream>>>((float4*)G1, M * (HID / 4));
    zero_f4<<<cdiv((size_t)M * NCLS / 4, T), T, 0, stream>>>((float4*)out, M * (NCLS / 4));
    deg_accum<<<cdiv(E, T), T, 0, stream>>>(dst, dinv, E);
    deg_rsqrt<<<cdiv(M, T), T, 0, stream>>>(dinv, M);

    // layer 1: H1 = x @ W1 (split-bf16 WMMA), edge scatter, epilogue+relu
    gemm_bf16x3<IN, HID><<<cdiv(M, 128), T, 0, stream>>>(x, W1, H1, M);
    agg_edges<HID><<<cdiv(E, 8), T, 0, stream>>>(H1, src, dst, dinv, G1, E);
    post1<<<cdiv((size_t)M * 32, T), T, 0, stream>>>(G1, H1, dinv, b1, M);

    // layer 2: H2 = h @ W2, edge scatter into d_out, epilogue (bias + self loop)
    gemm_bf16x3<HID, NCLS><<<cdiv(M, 128), T, 0, stream>>>(G1, W2, H2, M);
    agg_edges<NCLS><<<cdiv(E, 8), T, 0, stream>>>(H2, src, dst, dinv, out, E);
    post2<<<cdiv((size_t)M * 16, T), T, 0, stream>>>(out, H2, dinv, b2, M);
}
